// GraphormerAttention_79602923864118
// MI455X (gfx1250) — matrix operand order repaired and checked by reading, not compiled, not run on hardware
//
#include <hip/hip_runtime.h>

typedef __attribute__((ext_vector_type(16))) _Float16 v16h;
typedef __attribute__((ext_vector_type(8)))  _Float16 v8h;
typedef __attribute__((ext_vector_type(8)))  float    v8f;

#define DEV __device__ __forceinline__

DEV v8f wmma_f16(v16h a, v16h b, v8f c) {
  // 8 args: (neg_a, A, neg_b, B, c_mod, C, reuse_a, reuse_b)
  return __builtin_amdgcn_wmma_f32_16x16x32_f16(false, a, false, b, (short)0, c,
                                                false, false);
}

DEV v16h pack16(v8h lo, v8h hi) {
  v16h r;
#pragma unroll
  for (int i = 0; i < 8; ++i) { r[i] = lo[i]; r[8 + i] = hi[i]; }
  return r;
}

// A-operand (16x32, f16 WMMA): lane holds row (lane&15), two 8-half runs at
// k = h*8 and k = 16 + h*8  (h = lane>>4).
DEV v16h loadA_f16(const _Float16* base, int row, int ld, int kc, int hf) {
  const _Float16* p = base + (size_t)row * ld + kc;
  v8h lo = *(const v8h*)(p + hf * 8);
  v8h hi = *(const v8h*)(p + 16 + hf * 8);
  return pack16(lo, hi);
}
// B-operand (32x16): lane holds column (lane&15) == row of B-transposed,
// two 8-half runs at k = h*8 and k = 16 + h*8.  Source is row-major Bᵀ (n x K).
DEV v16h loadB_f16(const _Float16* base, int nrow, int ld, int kc, int hf) {
  const _Float16* p = base + (size_t)nrow * ld + kc;
  v8h lo = *(const v8h*)(p + hf * 8);
  v8h hi = *(const v8h*)(p + 16 + hf * 8);
  return pack16(lo, hi);
}

// --------------------------------------------------- f32 -> f16 convert ----
__global__ __launch_bounds__(256) void cvt_kernel(
    const float* __restrict__ src, _Float16* __restrict__ dst, int n8)
{
  int i = blockIdx.x * 256 + threadIdx.x;
  if (i < n8) {
    v8f s = *(const v8f*)(src + (size_t)i * 8);
    v8h d;
#pragma unroll
    for (int j = 0; j < 8; ++j) d[j] = (_Float16)s[j];
    *(v8h*)(dst + (size_t)i * 8) = d;
  }
}

// ---------------------------------------------------------------- QKV ------
// One wave per (16-row tile of B*N, 16-feature tile of 256). 8 k-chunks of 32.
__global__ __launch_bounds__(256) void qkv_kernel(
    const _Float16* __restrict__ xh,
    const _Float16* __restrict__ Wqh, const float* __restrict__ bq,
    const _Float16* __restrict__ Wkh, const float* __restrict__ bk,
    const _Float16* __restrict__ Wvh, const float* __restrict__ bv,
    _Float16* __restrict__ qws, _Float16* __restrict__ kws,
    _Float16* __restrict__ vtws)
{
  const int lane = threadIdx.x & 31;
  const int wv   = threadIdx.x >> 5;
  const int wid  = blockIdx.x * 8 + wv;   // 0..4095
  const int rt   = wid >> 4;              // token-tile 0..255
  const int ft   = wid & 15;              // feature-tile 0..15
  const int m    = lane & 15;
  const int hf   = lane >> 4;

  v8f aq = {}, ak = {}, av = {};
  const int arow = rt * 16 + m;           // flat token row (B*N)
  const int brow = ft * 16 + m;           // W output-feature row

#pragma unroll
  for (int kc = 0; kc < 8; ++kc) {
    v16h a   = loadA_f16(xh,  arow, 256, kc * 32, hf);
    v16h bqm = loadB_f16(Wqh, brow, 256, kc * 32, hf);
    v16h bkm = loadB_f16(Wkh, brow, 256, kc * 32, hf);
    v16h bvm = loadB_f16(Wvh, brow, 256, kc * 32, hf);
    aq = wmma_f16(a, bqm, aq);
    ak = wmma_f16(a, bkm, ak);
    av = wmma_f16(a, bvm, av);
  }

  const int f  = ft * 16 + m;             // global output feature
  const int hh = f >> 5, d = f & 31;      // head, dim-in-head
  const int b  = rt >> 6;                 // batch (64 tiles per batch)
  const int tokbase = (rt & 63) * 16;
  const float bqv = bq[f], bkv = bk[f], bvv = bv[f];
  const float scale = 0.17677669529663687f;   // 32^-0.5 folded into Q
  const size_t head = (size_t)(b * 8 + hh);

#pragma unroll
  for (int r = 0; r < 8; ++r) {
    int tok = tokbase + r + 8 * hf;       // D layout: m = r + 8*h
    size_t qi = (head * 1024 + tok) * 32 + d;
    qws[qi] = (_Float16)((aq[r] + bqv) * scale);
    kws[qi] = (_Float16)(ak[r] + bkv);
    vtws[(head * 32 + d) * 1024 + tok] = (_Float16)(av[r] + bvv);  // V transposed
  }
}

// ---------------------------------------------------- flash attention ------
// One wave per (b, head, 16-query tile); 32-key steps; online softmax.
__global__ __launch_bounds__(256) void attn_kernel(
    const float* __restrict__ z, const unsigned char* __restrict__ kmask,
    const float* __restrict__ z_emb,
    const _Float16* __restrict__ qws, const _Float16* __restrict__ kws,
    const _Float16* __restrict__ vtws, _Float16* __restrict__ tmpws)
{
  __shared__ __align__(16) _Float16 plds[8][16][32];   // per-wave P tile
  const int lane = threadIdx.x & 31;
  const int wv   = threadIdx.x >> 5;
  const int wid  = blockIdx.x * 8 + wv;   // 0..2047
  const int b    = wid >> 9;
  const int hh   = (wid >> 6) & 7;
  const int qt   = wid & 63;
  const int nn   = lane & 15;
  const int hf   = lane >> 4;

  // z_emb bin table replicated in both 16-lane halves (for ds_bpermute).
  const int tabbits = __float_as_int(z_emb[nn * 8 + hh]);

  const _Float16* qh = qws  + (size_t)(b * 8 + hh) * 1024 * 32;
  const _Float16* kh = kws  + (size_t)(b * 8 + hh) * 1024 * 32;
  const _Float16* vh = vtws + (size_t)(b * 8 + hh) * 32 * 1024;
  const float* zb = z + ((size_t)b * 1024 + qt * 16) * 1024;
  const unsigned char* mb = kmask + b * 1024;

  v16h aqv = loadA_f16(qh, qt * 16 + nn, 32, 0, hf);

  float mr[8], lr[8];
  v8f o0 = {}, o1 = {};
#pragma unroll
  for (int r = 0; r < 8; ++r) { mr[r] = -1e30f; lr[r] = 0.f; }

  for (int kt = 0; kt < 32; ++kt) {
    const int k0 = kt * 32;
    v16h bk0 = loadB_f16(kh, k0 + nn,      32, 0, hf);
    v16h bk1 = loadB_f16(kh, k0 + 16 + nn, 32, 0, hf);
    v8f zero = {};
    v8f s0 = wmma_f16(aqv, bk0, zero);     // S = Q*scale @ Kᵀ
    v8f s1 = wmma_f16(aqv, bk1, zero);

    // V B-operands: issue early so latency hides under softmax VALU work.
    v16h bv0 = loadB_f16(vh, nn,      1024, k0, hf);
    v16h bv1 = loadB_f16(vh, 16 + nn, 1024, k0, hf);

    const int key0 = k0 + nn, key1 = k0 + 16 + nn;
    const bool mk0 = mb[key0] != 0;
    const bool mk1 = mb[key1] != 0;

#pragma unroll
    for (int r = 0; r < 8; ++r) {          // add z-bin bias, apply mask
      const float* zr = zb + (size_t)(r + 8 * hf) * 1024;
      float z0 = zr[key0], z1 = zr[key1];
      int i0 = (int)(z0 / 5.0f * 16.0f); i0 = i0 < 0 ? 0 : (i0 > 15 ? 15 : i0);
      int i1 = (int)(z1 / 5.0f * 16.0f); i1 = i1 < 0 ? 0 : (i1 > 15 ? 15 : i1);
      float b0 = __int_as_float(__builtin_amdgcn_ds_bpermute(i0 << 2, tabbits));
      float b1 = __int_as_float(__builtin_amdgcn_ds_bpermute(i1 << 2, tabbits));
      s0[r] = mk0 ? -1e30f : s0[r] + b0;
      s1[r] = mk1 ? -1e30f : s1[r] + b1;
    }

#pragma unroll
    for (int r = 0; r < 8; ++r) {          // online softmax, row m = r+8*hf
      float rm = fmaxf(s0[r], s1[r]);
      rm = fmaxf(rm, __shfl_xor(rm, 1, 32));
      rm = fmaxf(rm, __shfl_xor(rm, 2, 32));
      rm = fmaxf(rm, __shfl_xor(rm, 4, 32));
      rm = fmaxf(rm, __shfl_xor(rm, 8, 32));
      float mnew = fmaxf(mr[r], rm);
      float corr = __expf(mr[r] - mnew);
      mr[r] = mnew;
      float p0 = __expf(s0[r] - mnew);
      float p1 = __expf(s1[r] - mnew);
      float rs = p0 + p1;
      rs += __shfl_xor(rs, 1, 32);
      rs += __shfl_xor(rs, 2, 32);
      rs += __shfl_xor(rs, 4, 32);
      rs += __shfl_xor(rs, 8, 32);
      lr[r] = lr[r] * corr + rs;
      o0[r] *= corr;
      o1[r] *= corr;
      plds[wv][r + 8 * hf][nn]      = (_Float16)p0;   // D layout -> LDS
      plds[wv][r + 8 * hf][16 + nn] = (_Float16)p1;
    }

    // reload P in A layout (wave-private LDS, same-wave DS ops are in-order)
    v16h pa = loadA_f16(&plds[wv][0][0], nn, 32, 0, hf);
    o0 = wmma_f16(pa, bv0, o0);
    o1 = wmma_f16(pa, bv1, o1);
  }

#pragma unroll
  for (int r = 0; r < 8; ++r) {
    float inv = lr[r] > 0.f ? 1.0f / lr[r] : 0.f;
    int tok = qt * 16 + r + 8 * hf;
    size_t o = ((size_t)b * 1024 + tok) * 256 + hh * 32;
    tmpws[o + nn]      = (_Float16)(o0[r] * inv);
    tmpws[o + 16 + nn] = (_Float16)(o1[r] * inv);
  }
}

// ------------------------------------------------------- out projection ----
__global__ __launch_bounds__(256) void proj_kernel(
    const _Float16* __restrict__ tmpws,
    const _Float16* __restrict__ Woh, const float* __restrict__ bo,
    float* __restrict__ out)
{
  const int lane = threadIdx.x & 31;
  const int wv   = threadIdx.x >> 5;
  const int wid  = blockIdx.x * 8 + wv;   // 0..4095
  const int rt   = wid >> 4;
  const int ft   = wid & 15;
  const int m    = lane & 15;
  const int hf   = lane >> 4;

  v8f acc = {};
  const int arow = rt * 16 + m;
  const int brow = ft * 16 + m;
#pragma unroll
  for (int kc = 0; kc < 8; ++kc) {
    v16h a  = loadA_f16(tmpws, arow, 256, kc * 32, hf);
    v16h bm = loadB_f16(Woh,   brow, 256, kc * 32, hf);
    acc = wmma_f16(a, bm, acc);
  }
  const int f = ft * 16 + m;
  const float bov = bo[f];
#pragma unroll
  for (int r = 0; r < 8; ++r)
    out[((size_t)(rt * 16 + r + 8 * hf)) * 256 + f] = acc[r] + bov;
}

extern "C" void kernel_launch(void* const* d_in, const int* in_sizes, int n_in,
                              void* d_out, int out_size, void* d_ws, size_t ws_size,
                              hipStream_t stream) {
  const float*         x     = (const float*)d_in[0];
  const float*         z     = (const float*)d_in[1];
  const unsigned char* kmask = (const unsigned char*)d_in[2];
  const float* Wq = (const float*)d_in[3];
  const float* bq = (const float*)d_in[4];
  const float* Wk = (const float*)d_in[5];
  const float* bk = (const float*)d_in[6];
  const float* Wv = (const float*)d_in[7];
  const float* bv = (const float*)d_in[8];
  const float* Wo = (const float*)d_in[9];
  const float* bo = (const float*)d_in[10];
  const float* z_emb = (const float*)d_in[11];

  const size_t NX = (size_t)4 * 1024 * 256;       // x elements   (1M)
  const size_t NW = (size_t)256 * 256;            // W elements   (64K)
  const size_t E  = (size_t)4 * 8 * 1024 * 32;    // per Q/K/V/tmp buffer (1M)

  _Float16* xh    = (_Float16*)d_ws;              // 2 MB
  _Float16* Wqh   = xh + NX;                      // 128 KB each
  _Float16* Wkh   = Wqh + NW;
  _Float16* Wvh   = Wkh + NW;
  _Float16* Woh   = Wvh + NW;
  _Float16* qws   = Woh + NW;                     // 2 MB each
  _Float16* kws   = qws + E;
  _Float16* vtws  = kws + E;
  _Float16* tmpws = vtws + E;                     // total ~10.5 MB of ws

  cvt_kernel<<<(int)(NX / 8 / 256), 256, 0, stream>>>(x,  xh,  (int)(NX / 8));
  cvt_kernel<<<(int)(NW / 8 / 256), 256, 0, stream>>>(Wq, Wqh, (int)(NW / 8));
  cvt_kernel<<<(int)(NW / 8 / 256), 256, 0, stream>>>(Wk, Wkh, (int)(NW / 8));
  cvt_kernel<<<(int)(NW / 8 / 256), 256, 0, stream>>>(Wv, Wvh, (int)(NW / 8));
  cvt_kernel<<<(int)(NW / 8 / 256), 256, 0, stream>>>(Wo, Woh, (int)(NW / 8));

  qkv_kernel<<<512, 256, 0, stream>>>(xh, Wqh, bq, Wkh, bk, Wvh, bv,
                                      qws, kws, vtws);
  attn_kernel<<<256, 256, 0, stream>>>(z, kmask, z_emb, qws, kws, vtws, tmpws);
  proj_kernel<<<512, 256, 0, stream>>>(tmpws, Woh, bo, (float*)d_out);
}
